// add_transformer_48713519071868
// MI455X (gfx1250) — matrix operand order repaired and checked by reading, not compiled, not run on hardware
//
#include <hip/hip_runtime.h>

typedef _Float16 v16h __attribute__((ext_vector_type(16)));
typedef _Float16 v8h  __attribute__((ext_vector_type(8)));
typedef float    v8f  __attribute__((ext_vector_type(8)));

#define D_MODEL 512
#define LQ      4096
#define LK      5119
#define WIN     1024
#define NT_CNT  32   // 512 / 16 N-tiles
#define KT_CNT  16   // 512 / 32 K-steps

// ---------------------------------------------------------------------------
// Repack f32 weight [K=512, N=512] row-major into WMMA B-operand f16 layout:
// tile (kt, nt): 32 lanes x 16 f16, lane<16 -> col=nt*16+lane, K=kt*32+j
//                lane>=16 -> col=nt*16+(lane-16), K=kt*32+16+j
// ---------------------------------------------------------------------------
__global__ void pack_w_kernel(const float* __restrict__ W, _Float16* __restrict__ Wp) {
    int idx = blockIdx.x * blockDim.x + threadIdx.x;   // tile*32 + lane
    if (idx >= KT_CNT * NT_CNT * 32) return;
    int lane = idx & 31;
    int tile = idx >> 5;
    int nt = tile & (NT_CNT - 1);
    int kt = tile >> 5;
    int col   = nt * 16 + (lane & 15);
    int kbase = kt * 32 + (lane >> 4) * 16;
    _Float16* dst = Wp + (size_t)idx * 16;
#pragma unroll
    for (int j = 0; j < 16; ++j)
        dst[j] = (_Float16)W[(size_t)(kbase + j) * D_MODEL + col];
}

// ---------------------------------------------------------------------------
// Fused LayerNorm + projection GEMM. 16 rows per workgroup, 256 threads
// (8 waves). DUAL=true computes two projections (k and v) on the same rows
// and writes kv = relu(kp)*vp; DUAL=false writes relu(qp).
// LDS sized per-instantiation: 16KB (single) / 32KB (dual) for occupancy.
// ---------------------------------------------------------------------------
template <bool DUAL>
__global__ void ln_proj_kernel(const float* __restrict__ X1, const float* __restrict__ X2,
                               const float* __restrict__ g1, const float* __restrict__ b1,
                               const float* __restrict__ g2, const float* __restrict__ b2,
                               const _Float16* __restrict__ W1p, const _Float16* __restrict__ W2p,
                               const float* __restrict__ bw1, const float* __restrict__ bw2,
                               float* __restrict__ Out, int L) {
    __shared__ __align__(16) _Float16 ldsBuf[(DUAL ? 32 : 16) * D_MODEL];
    _Float16* ldsA = ldsBuf;
    _Float16* ldsV = DUAL ? (ldsBuf + 16 * D_MODEL) : ldsBuf;

    const int tilesPer = (L + 15) >> 4;
    const int b    = blockIdx.x / tilesPer;
    const int row0 = (blockIdx.x % tilesPer) * 16;

    // ---- LayerNorm phase: 16 threads per row, 32 elements per thread ----
    {
        const int r = threadIdx.x >> 4;     // row in tile 0..15
        const int j = threadIdx.x & 15;     // sub-thread 0..15
        int row = row0 + r;
        int rowc = row < L ? row : L - 1;
        for (int pass = 0; pass < (DUAL ? 2 : 1); ++pass) {
            const float* X  = pass ? X2 : X1;
            const float* g  = pass ? g2 : g1;
            const float* bb = pass ? b2 : b1;
            _Float16* lds   = pass ? ldsV : ldsA;
            const float* xp = X + ((size_t)b * L + rowc) * D_MODEL;
            float x[32];
            float s = 0.f;
#pragma unroll
            for (int t = 0; t < 32; ++t) { x[t] = xp[j + t * 16]; s += x[t]; }
#pragma unroll
            for (int o = 8; o; o >>= 1) s += __shfl_xor(s, o, 32);
            float mean = s * (1.f / 512.f);
            float vs = 0.f;
#pragma unroll
            for (int t = 0; t < 32; ++t) { float d = x[t] - mean; vs += d * d; }
#pragma unroll
            for (int o = 8; o; o >>= 1) vs += __shfl_xor(vs, o, 32);
            float inv = rsqrtf(vs * (1.f / 512.f) + 1e-5f);
#pragma unroll
            for (int t = 0; t < 32; ++t) {
                int e = j + t * 16;
                float y = (x[t] - mean) * inv * g[e] + bb[e];
                lds[r * D_MODEL + e] = (_Float16)y;
            }
        }
    }
    __syncthreads();

    // ---- WMMA GEMM phase: each wave owns 4 N-tiles ----
    const int wave = threadIdx.x >> 5;
    const int lane = threadIdx.x & 31;
    const int lrow = lane & 15;
    const int kSel = lane >> 4;            // half-wave select

    v8f acc1[4] = {v8f{}, v8f{}, v8f{}, v8f{}};
    v8f acc2[4] = {v8f{}, v8f{}, v8f{}, v8f{}};

    for (int kt = 0; kt < KT_CNT; ++kt) {
        const int kA = kt * 32 + kSel * 16;
        // A operand: two contiguous 8-f16 runs per lane from LDS
        v8h lo1 = *(const v8h*)&ldsA[lrow * D_MODEL + kA];
        v8h hi1 = *(const v8h*)&ldsA[lrow * D_MODEL + kA + 8];
        v16h a1 = __builtin_shufflevector(lo1, hi1, 0,1,2,3,4,5,6,7,8,9,10,11,12,13,14,15);
        v16h a2;
        if (DUAL) {
            v8h lo2 = *(const v8h*)&ldsV[lrow * D_MODEL + kA];
            v8h hi2 = *(const v8h*)&ldsV[lrow * D_MODEL + kA + 8];
            a2 = __builtin_shufflevector(lo2, hi2, 0,1,2,3,4,5,6,7,8,9,10,11,12,13,14,15);
        }
#pragma unroll
        for (int q = 0; q < 4; ++q) {
            const int nt = wave * 4 + q;
            const size_t boff = (((size_t)kt * NT_CNT + nt) * 32 + lane) * 16;
            v16h bm1 = *(const v16h*)(W1p + boff);
            acc1[q] = __builtin_amdgcn_wmma_f32_16x16x32_f16(
                false, a1, false, bm1, (short)0, acc1[q], false, false);
            if (DUAL) {
                v16h bm2 = *(const v16h*)(W2p + boff);
                acc2[q] = __builtin_amdgcn_wmma_f32_16x16x32_f16(
                    false, a2, false, bm2, (short)0, acc2[q], false, false);
            }
        }
    }

    // ---- Store: C layout VGPR r -> M = kSel*8 + r, N = nt*16 + lrow ----
#pragma unroll
    for (int q = 0; q < 4; ++q) {
        const int nt  = wave * 4 + q;
        const int col = nt * 16 + lrow;
        const float bv1 = bw1[col];
        const float bv2 = DUAL ? bw2[col] : 0.f;
#pragma unroll
        for (int r = 0; r < 8; ++r) {
            int row = row0 + kSel * 8 + r;
            if (row < L) {
                float p1 = acc1[q][r] + bv1;
                float* op = Out + ((size_t)b * L + row) * D_MODEL + col;
                if (DUAL) {
                    float p2 = acc2[q][r] + bv2;
                    *op = fmaxf(p1, 0.f) * p2;      // kv = relu(kp) * vp
                } else {
                    *op = fmaxf(p1, 0.f);           // relu(qp)
                }
            }
        }
    }
}

// ---------------------------------------------------------------------------
// Sliding-window sum (window=1024, sup=0) + multiply by relu(qp).
// Block: 256 threads = 256 contiguous d-columns; each block owns a 128-row
// chunk. Initial window built once, then +/- recurrence (kv lives in L2).
// ---------------------------------------------------------------------------
__global__ void window_kernel(const float* __restrict__ qp,
                              const float* __restrict__ kv,
                              float* __restrict__ out) {
    const int RC = 128;
    const int chunks = LQ / RC;            // 32
    int bid = blockIdx.x;
    int dh    = bid & 1;
    int chunk = (bid >> 1) % chunks;
    int b     = (bid >> 1) / chunks;
    int col   = dh * 256 + threadIdx.x;

    const float* kvp = kv + (size_t)b * LK * D_MODEL + col;
    const float* qpp = qp + (size_t)b * LQ * D_MODEL + col;
    float*       op  = out + (size_t)b * LQ * D_MODEL + col;

    const int i0 = chunk * RC;
    float w = 0.f;
    for (int j = 0; j < WIN; ++j)
        w += kvp[(size_t)(i0 + j) * D_MODEL];

    for (int i = i0; i < i0 + RC; ++i) {
        op[(size_t)i * D_MODEL] = qpp[(size_t)i * D_MODEL] * w;
        if (i + 1 < i0 + RC)
            w += kvp[(size_t)(i + WIN) * D_MODEL] - kvp[(size_t)i * D_MODEL];
    }
}

// ---------------------------------------------------------------------------
extern "C" void kernel_launch(void* const* d_in, const int* in_sizes, int n_in,
                              void* d_out, int out_size, void* d_ws, size_t ws_size,
                              hipStream_t stream) {
    const float* q   = (const float*)d_in[0];
    const float* k   = (const float*)d_in[1];
    const float* v   = (const float*)d_in[2];
    const float* gq  = (const float*)d_in[3];
    const float* bq  = (const float*)d_in[4];
    const float* gk  = (const float*)d_in[5];
    const float* bk  = (const float*)d_in[6];
    const float* gv  = (const float*)d_in[7];
    const float* bv  = (const float*)d_in[8];
    const float* Wq  = (const float*)d_in[9];
    const float* bwq = (const float*)d_in[10];
    const float* Wk  = (const float*)d_in[11];
    const float* bwk = (const float*)d_in[12];
    const float* Wv  = (const float*)d_in[13];
    const float* bwv = (const float*)d_in[14];
    float* out = (float*)d_out;

    // workspace layout
    char* ws = (char*)d_ws;
    const size_t qp_bytes = (size_t)4 * LQ * D_MODEL * sizeof(float);     // 33.5 MB
    const size_t kv_bytes = (size_t)4 * LK * D_MODEL * sizeof(float);     // 41.9 MB
    const size_t wp_bytes = (size_t)D_MODEL * D_MODEL * sizeof(_Float16); // 512 KB
    float*    qp_ws = (float*)ws;
    float*    kv_ws = (float*)(ws + qp_bytes);
    _Float16* wq_p  = (_Float16*)(ws + qp_bytes + kv_bytes);
    _Float16* wk_p  = (_Float16*)(ws + qp_bytes + kv_bytes + wp_bytes);
    _Float16* wv_p  = (_Float16*)(ws + qp_bytes + kv_bytes + 2 * wp_bytes);

    // 1) repack weights to f16 WMMA B-layout (16384 threads each)
    pack_w_kernel<<<64, 256, 0, stream>>>(Wq, wq_p);
    pack_w_kernel<<<64, 256, 0, stream>>>(Wk, wk_p);
    pack_w_kernel<<<64, 256, 0, stream>>>(Wv, wv_p);

    // 2) relu(LN(q)@Wq + bwq) -> qp_ws
    {
        int tiles = (LQ + 15) / 16;              // 256
        ln_proj_kernel<false><<<4 * tiles, 256, 0, stream>>>(
            q, nullptr, gq, bq, nullptr, nullptr,
            wq_p, nullptr, bwq, nullptr, qp_ws, LQ);
    }
    // 3) kv = relu(LN(k)@Wk + bwk) * (LN(v)@Wv + bwv) -> kv_ws
    {
        int tiles = (LK + 15) / 16;              // 320
        ln_proj_kernel<true><<<4 * tiles, 256, 0, stream>>>(
            k, v, gk, bk, gv, bv,
            wk_p, wv_p, bwk, bwv, kv_ws, LK);
    }
    // 4) out = qp_relu * sliding_window_sum(kv)
    window_kernel<<<4 * (LQ / 128) * 2, 256, 0, stream>>>(qp_ws, kv_ws, out);
}
